// BinaryClassifier_63848983823194
// MI455X (gfx1250) — compile-verified, matrix operand order checked by reading, not run on hardware
//
#include <hip/hip_runtime.h>
#include <hip/hip_bf16.h>

typedef __attribute__((ext_vector_type(16))) _Float16 v16h;
typedef __attribute__((ext_vector_type(8)))  _Float16 v8h;
typedef __attribute__((ext_vector_type(8)))  float    v8f;
typedef __attribute__((ext_vector_type(4)))  float    v4f;

#define SPB 16      // samples per block
#define THREADS 128 // 4 waves (wave32)

// K-index mapping for 16-bit A/B WMMA fragments (16x32 tile, wave32):
//  pair p = h/2:  p<4 -> K = 2p, p>=4 -> K = 2p+8;  lanes 16..31 add +8; +chunk*32; +h&1
__device__ __forceinline__ int kmap(int chunk, int h, int laneHi) {
    int p = h >> 1, lo = h & 1;
    return 32 * chunk + 2 * p + ((p >= 4) ? 8 : 0) + (laneHi ? 8 : 0) + lo;
}

__global__ __launch_bounds__(THREADS)
void qcnn_fused(const float* __restrict__ x,
                const float* __restrict__ c1w, const float* __restrict__ c1b,
                const float* __restrict__ c2w, const float* __restrict__ c2b,
                const float* __restrict__ f1w, const float* __restrict__ f1b,
                const float* __restrict__ f2w, const float* __restrict__ f2b,
                const float* __restrict__ f3w, const float* __restrict__ f3b,
                const float* __restrict__ qpp,
                float* __restrict__ out)
{
    // Union blob: [stages A-B] f32 input tile (50176 B)
    //             [stages C-E] a1 (8192 B) | csc (4096 B) | h1 (4096 B)
    __shared__ __align__(16) unsigned char smem_u[SPB * 784 * 4];
    float*    in_f = (float*)smem_u;                  // SPB x 784 f32
    _Float16* a1   = (_Float16*)smem_u;               // SPB x 256 f16 (fc1 input)
    float*    csc  = (float*)(smem_u + 8192);         // 4 waves x 256 f32 C scratch
    float*    h1   = (float*)(smem_u + 12288);        // SPB x 64 f32 (fc1 output)

    __shared__ _Float16 c1p[SPB * 288];               // pooled conv1: [s][ch][12][12]
    __shared__ __align__(16) _Float16 w2h[16 * 64];   // conv2 weights f16, K 50->64
    __shared__ float w1c[50], b1c[2], b2c[16], f1bc[64], f2wc[128], cst[12];
    // cst: [0..1]=fc2_b, [2]=fc3_w, [3]=fc3_b, [4..11]=qnn_params

    const int t      = threadIdx.x;
    const int lane   = t & 31;
    const int wave   = t >> 5;
    const int laneHi = (lane >= 16) ? 1 : 0;
    const long base  = (long)blockIdx.x * (SPB * 784);

    // ---------------- Stage A: async-DMA input tile into LDS --------------
    // GLOBAL_LOAD_ASYNC_TO_LDS_B128: memory -> LDS without touching VGPRs.
    {
        const unsigned long long gbase = (unsigned long long)(x + base);
        for (int i = t; i < (SPB * 784) / 4; i += THREADS) {
            unsigned ldsa = (unsigned)(size_t)(in_f + i * 4);      // LDS byte offset
            unsigned long long ga = gbase + (unsigned long long)i * 16ull;
            asm volatile("global_load_async_to_lds_b128 %0, %1, off"
                         :: "v"(ldsa), "v"(ga) : "memory");
        }
    }
    if (t == 0) __builtin_prefetch(x + base + SPB * 784, 0, 1);    // next tile

    if (t < 50) w1c[t] = c1w[t];
    if (t < 2)  b1c[t] = c1b[t];
    if (t < 16) b2c[t] = c2b[t];
    if (t < 64) f1bc[t] = f1b[t];
    for (int i = t; i < 128; i += THREADS) f2wc[i] = f2w[i];
    if (t < 2)            cst[t] = f2b[t];
    else if (t == 2)      cst[2] = f3w[0];
    else if (t == 3)      cst[3] = f3b[0];
    else if (t >= 4 && t < 12) cst[t] = qpp[t - 4];
    for (int i = t; i < 16 * 64; i += THREADS) {
        int ch = i >> 6, k = i & 63;
        w2h[i] = (k < 50) ? (_Float16)c2w[ch * 50 + k] : (_Float16)0.f;
    }
    asm volatile("s_wait_asynccnt 0" ::: "memory");  // my async copies landed
    __syncthreads();                                 // everyone's copies visible

    // ---------------- Stage B: conv1 + bias + relu + 2x2 maxpool (VALU) ---
    for (int o = t; o < SPB * 288; o += THREADS) {
        int s = o / 288, rem = o % 288;
        int ch = rem / 144, ij = rem % 144;
        int i = ij / 12, j = ij % 12;
        const float* ip = in_f + s * 784;
        const float* w = &w1c[ch * 25];
        float mx = -1e30f;
        #pragma unroll
        for (int di = 0; di < 2; ++di) {
            #pragma unroll
            for (int dj = 0; dj < 2; ++dj) {
                int r0 = 2 * i + di, c0 = 2 * j + dj;
                float acc = 0.f;
                #pragma unroll
                for (int u = 0; u < 5; ++u)
                    #pragma unroll
                    for (int v = 0; v < 5; ++v)
                        acc += ip[(r0 + u) * 28 + (c0 + v)] * w[u * 5 + v];
                mx = fmaxf(mx, acc);
            }
        }
        c1p[o] = (_Float16)fmaxf(mx + b1c[ch], 0.f);
    }
    __syncthreads();   // in_f dead from here; union region becomes a1/csc/h1

    // ---------------- Stage C: conv2 as im2col WMMA + bias + relu + pool --
    // B fragments (16 output channels as columns): contiguous -> ds_load_b128
    v16h b2f[2];
    {
        int n = lane & 15;
        #pragma unroll
        for (int c = 0; c < 2; ++c) {
            int kb = 32 * c + 8 * laneHi;
            v8h blo = *(const v8h*)(w2h + n * 64 + kb);
            v8h bhi = *(const v8h*)(w2h + n * 64 + kb + 16);
            #pragma unroll
            for (int h = 0; h < 8; ++h) { b2f[c][h] = blo[h]; b2f[c][h + 8] = bhi[h]; }
        }
    }
    for (int T = wave * 16; T < wave * 16 + 16; ++T) {   // 64 row-tiles / 4 waves
        int s = T >> 2, ti = T & 3;                      // sample, i-pair tile
        int m = lane & 15;
        int p = ti * 16 + m;                             // output position 0..63
        int ii = p >> 3, jj = p & 7;
        v8f acc = {};
        #pragma unroll
        for (int c = 0; c < 2; ++c) {
            v16h af;                                     // scattered im2col gather
            #pragma unroll
            for (int h = 0; h < 16; ++h) {
                int k = kmap(c, h, laneHi);
                _Float16 val = (_Float16)0.f;
                if (k < 50) {
                    int cin = (k >= 25) ? 1 : 0;
                    int r2 = k - cin * 25;
                    int u = r2 / 5, vv = r2 % 5;
                    val = c1p[s * 288 + cin * 144 + (ii + u) * 12 + (jj + vv)];
                }
                af[h] = val;
            }
            acc = __builtin_amdgcn_wmma_f32_16x16x32_f16(
                    false, af, false, b2f[c], (short)0, acc, false, false);
        }
        // dump C tile into wave-private scratch; LDS ops of a wave are in-order,
        // so a wave-local dscnt wait replaces a block barrier here.
        float* cs = csc + wave * 256;
        #pragma unroll
        for (int r = 0; r < 8; ++r)
            cs[(r + 8 * laneHi) * 16 + (lane & 15)] = acc[r];
        asm volatile("s_wait_dscnt 0" ::: "memory");
        #pragma unroll
        for (int q = 0; q < 2; ++q) {
            int idx = lane + q * 32;              // 64 pooled values per tile
            int ch = idx & 15, jp = idx >> 4;     // channel, pooled j
            float m0 = -1e30f;
            #pragma unroll
            for (int di = 0; di < 2; ++di)
                #pragma unroll
                for (int dj = 0; dj < 2; ++dj)
                    m0 = fmaxf(m0, cs[(di * 8 + 2 * jp + dj) * 16 + ch]);
            a1[s * 256 + ch * 16 + ti * 4 + jp] =
                (_Float16)fmaxf(m0 + b2c[ch], 0.f);
        }
    }
    __syncthreads();

    // ---------------- Stage D: fc1 (16x256 @ 256x64) via WMMA -------------
    {
        int nt = wave;                    // one 16-col N-tile per wave
        int mrow = lane & 15;
        const float* wrow = f1w + (nt * 16 + mrow) * 256;   // B column in f1w
        v8f acc = {};
        #pragma unroll
        for (int c = 0; c < 8; ++c) {     // K = 256 -> 8 chunks of 32
            int kb = 32 * c + 8 * laneHi;
            v8h alo = *(const v8h*)(a1 + mrow * 256 + kb);
            v8h ahi = *(const v8h*)(a1 + mrow * 256 + kb + 16);
            v4f w0 = *(const v4f*)(wrow + kb);
            v4f w1 = *(const v4f*)(wrow + kb + 4);
            v4f w2 = *(const v4f*)(wrow + kb + 16);
            v4f w3 = *(const v4f*)(wrow + kb + 20);
            v16h af, bf;
            #pragma unroll
            for (int h = 0; h < 8; ++h) { af[h] = alo[h]; af[h + 8] = ahi[h]; }
            #pragma unroll
            for (int h = 0; h < 4; ++h) {
                bf[h]      = (_Float16)w0[h];
                bf[h + 4]  = (_Float16)w1[h];
                bf[h + 8]  = (_Float16)w2[h];
                bf[h + 12] = (_Float16)w3[h];
            }
            acc = __builtin_amdgcn_wmma_f32_16x16x32_f16(
                    false, af, false, bf, (short)0, acc, false, false);
        }
        #pragma unroll
        for (int r = 0; r < 8; ++r) {
            int M = r + 8 * laneHi, N = nt * 16 + (lane & 15);
            h1[M * 64 + N] = fmaxf(acc[r] + f1bc[N], 0.f);
        }
    }
    __syncthreads();

    // ---------------- Stage E: fc2 + quantum circuit + fc3 + log_softmax --
    if (t < SPB) {
        int s = t;
        float h2[2];
        #pragma unroll
        for (int c = 0; c < 2; ++c) {
            float acc = cst[c];
            for (int k = 0; k < 64; ++k)
                acc += h1[s * 64 + k] * f2wc[c * 64 + k];
            h2[c] = acc;
        }
        float x0 = h2[0], x1 = h2[1];

        float ar[2][2], ai[2][2];
        float ph00 = -x0 - x1, ph01 = -x0 + x1, ph10 = x0 - x1, ph11 = x0 + x1;
        ar[0][0] = 0.5f * cosf(ph00); ai[0][0] = 0.5f * sinf(ph00);
        ar[0][1] = 0.5f * cosf(ph01); ai[0][1] = 0.5f * sinf(ph01);
        ar[1][0] = 0.5f * cosf(ph10); ai[1][0] = 0.5f * sinf(ph10);
        ar[1][1] = 0.5f * cosf(ph11); ai[1][1] = 0.5f * sinf(ph11);

        auto cnot01 = [&]() { float tr = ar[1][0], tm = ai[1][0];
            ar[1][0] = ar[1][1]; ai[1][0] = ai[1][1]; ar[1][1] = tr; ai[1][1] = tm; };
        auto cnot10 = [&]() { float tr = ar[0][1], tm = ai[0][1];
            ar[0][1] = ar[1][1]; ai[0][1] = ai[1][1]; ar[1][1] = tr; ai[1][1] = tm; };
        auto rot = [](float& re, float& im, float c, float sn) {
            float nr = re * c - im * sn; im = re * sn + im * c; re = nr; };
        auto ry0 = [&](float c, float sn) {
            #pragma unroll
            for (int k = 0; k < 2; ++k) {
                float r0 = ar[0][k], i0 = ai[0][k], r1 = ar[1][k], i1 = ai[1][k];
                ar[0][k] = c * r0 - sn * r1; ai[0][k] = c * i0 - sn * i1;
                ar[1][k] = sn * r0 + c * r1; ai[1][k] = sn * i0 + c * i1;
            } };
        auto ry1 = [&](float c, float sn) {
            #pragma unroll
            for (int j = 0; j < 2; ++j) {
                float r0 = ar[j][0], i0 = ai[j][0], r1 = ar[j][1], i1 = ai[j][1];
                ar[j][0] = c * r0 - sn * r1; ai[j][0] = c * i0 - sn * i1;
                ar[j][1] = sn * r0 + c * r1; ai[j][1] = sn * i0 + c * i1;
            } };

        cnot01();
        const float PI = 3.14159265358979323846f;
        float ang = (PI - x0) * (PI - x1);
        float ca = cosf(ang), sa = sinf(ang);
        rot(ar[0][0], ai[0][0], ca, -sa); rot(ar[1][0], ai[1][0], ca, -sa);
        rot(ar[0][1], ai[0][1], ca,  sa); rot(ar[1][1], ai[1][1], ca,  sa);
        cnot01();
        ry0(cosf(cst[4]), sinf(cst[4]));  ry1(cosf(cst[5]), sinf(cst[5]));
        cnot01();
        ry0(cosf(cst[6]), sinf(cst[6]));  ry1(cosf(cst[7]), sinf(cst[7]));
        cnot10();
        ry0(cosf(cst[8]), sinf(cst[8]));  ry1(cosf(cst[9]), sinf(cst[9]));
        cnot01();
        ry0(cosf(cst[10]), sinf(cst[10])); ry1(cosf(cst[11]), sinf(cst[11]));

        float q = (ar[0][0]*ar[0][0] + ai[0][0]*ai[0][0])
                - (ar[0][1]*ar[0][1] + ai[0][1]*ai[0][1])
                - (ar[1][0]*ar[1][0] + ai[1][0]*ai[1][0])
                + (ar[1][1]*ar[1][1] + ai[1][1]*ai[1][1]);

        float y  = q * cst[2] + cst[3];
        float l0 = y, l1 = 1.f - y;
        float mx = fmaxf(l0, l1);
        float lse = mx + logf(expf(l0 - mx) + expf(l1 - mx));
        long ob = ((long)blockIdx.x * SPB + s) * 2;
        out[ob]     = l0 - lse;
        out[ob + 1] = l1 - lse;
    }
}

extern "C" void kernel_launch(void* const* d_in, const int* in_sizes, int n_in,
                              void* d_out, int out_size, void* d_ws, size_t ws_size,
                              hipStream_t stream) {
    (void)n_in; (void)d_ws; (void)ws_size; (void)out_size;
    const float* x   = (const float*)d_in[0];
    const float* c1w = (const float*)d_in[1];
    const float* c1b = (const float*)d_in[2];
    const float* c2w = (const float*)d_in[3];
    const float* c2b = (const float*)d_in[4];
    const float* f1w = (const float*)d_in[5];
    const float* f1b = (const float*)d_in[6];
    const float* f2w = (const float*)d_in[7];
    const float* f2b = (const float*)d_in[8];
    const float* f3w = (const float*)d_in[9];
    const float* f3b = (const float*)d_in[10];
    const float* qp  = (const float*)d_in[11];
    float* out = (float*)d_out;

    int B = in_sizes[0] / 784;
    int grid = (B + SPB - 1) / SPB;   // B=65536 -> 4096 blocks
    qcnn_fused<<<grid, THREADS, 0, stream>>>(x, c1w, c1b, c2w, c2b,
                                             f1w, f1b, f2w, f2b, f3w, f3b,
                                             qp, out);
}
